// mha_65910568124786
// MI455X (gfx1250) — matrix operand order repaired and checked by reading, not compiled, not run on hardware
//
#include <hip/hip_runtime.h>

// ---------------------------------------------------------------------------
// MHA forward, B=8 S=1024 D=768 H=12 hd=64, fp32 in/out.
// bf16 WMMA (f32 accum) for all matmuls; flash-attention fusion so the 402MB
// S*S matrix never touches HBM. GEMM k-loops are software-pipelined (register
// double buffering, unrolled x2 so buffer rotation costs no v_mov);
// flash loop prefetches the next K/V chunk.
// ---------------------------------------------------------------------------

typedef __bf16 bf16_t;
typedef __attribute__((ext_vector_type(16))) __bf16 v16bf;
typedef __attribute__((ext_vector_type(8)))  __bf16 v8bf;
typedef __attribute__((ext_vector_type(8)))  float  v8f;

#define B_   8
#define S_   1024
#define D_   768
#define H_   12
#define HD_  64
#define M_   (B_ * S_)      // 8192
#define N3D_ (3 * D_)       // 2304

// round-to-nearest-even fp32 -> bf16
__device__ __forceinline__ bf16_t f2bf(float f) {
  unsigned u = __builtin_bit_cast(unsigned, f);
  unsigned r = u + 0x7FFFu + ((u >> 16) & 1u);
  unsigned short h = (unsigned short)(r >> 16);
  return __builtin_bit_cast(bf16_t, h);
}

__device__ __forceinline__ v8f wmma_bf16(v16bf a, v16bf b, v8f c) {
  return __builtin_amdgcn_wmma_f32_16x16x32_bf16(
      /*neg_a=*/false, a, /*neg_b=*/false, b,
      /*c_mod=*/(short)0, c, /*reuse_a=*/false, /*reuse_b=*/false);
}

// A fragment (16x32, M x K) from row-major [*, ld] matrix, rows row0..row0+15,
// k in [k0, k0+32). Layout used here: lane l -> m=l%16, g=l/16;
// elem i <-> k = k0+16g+i (contiguous).
__device__ __forceinline__ v16bf load_a_frag(const bf16_t* p, int ld,
                                             int row0, int k0, int lane) {
  int m = lane & 15, g = lane >> 4;
  return *reinterpret_cast<const v16bf*>(p + (size_t)(row0 + m) * ld +
                                         (k0 + 16 * g));
}

// B fragment (32x16, K x N) where the source is stored n-major: src[n][k],
// row stride ld. Covers n in [n0,n0+16), k in [k0,k0+32).
// Layout used here: lane l -> n=l%16, g=l/16; elem i <-> k = k0+16g+i (contiguous).
__device__ __forceinline__ v16bf load_b_frag(const bf16_t* p, int ld,
                                             int n0, int k0, int lane) {
  int n = lane & 15, g = lane >> 4;
  return *reinterpret_cast<const v16bf*>(p + (size_t)(n0 + n) * ld +
                                         (k0 + 16 * g));
}

// ---------------------------------------------------------------------------
__global__ void cvt_f32_bf16(const float* __restrict__ in,
                             bf16_t* __restrict__ out, int n) {
  int i = blockIdx.x * blockDim.x + threadIdx.x;
  int stride = gridDim.x * blockDim.x;
  for (; i < n; i += stride) out[i] = f2bf(in[i]);
}

// ---------------------------------------------------------------------------
// qkv = x @ w_qkv^T + b_qkv ; scatter into Q[B,H,S,hd], K[B,H,S,hd],
// Vt[B,H,hd,S] (V transposed so PV B-frags are contiguous later).
// Block 256 thr = 8 waves (4 M-waves x 2 N-waves); wave tile 32x64;
// WG tile 128(M) x 128(N). Grid (2304/128=18, 8192/128=64).
// k-loop software-pipelined; unroll x2 removes buffer-rotation moves.
__global__ __launch_bounds__(256) void qkv_gemm(
    const bf16_t* __restrict__ xb,   // [M_, D_]
    const bf16_t* __restrict__ wb,   // [N3D_, D_]
    const float* __restrict__ bias,  // [N3D_]
    bf16_t* __restrict__ Q, bf16_t* __restrict__ Kc, bf16_t* __restrict__ Vt) {
  const int lane = threadIdx.x & 31;
  const int wave = threadIdx.x >> 5;
  const int m0 = blockIdx.y * 128 + (wave & 3) * 32;
  const int n0 = blockIdx.x * 128 + (wave >> 2) * 64;

  v8f acc[2][4];
#pragma unroll
  for (int mt = 0; mt < 2; ++mt)
#pragma unroll
    for (int t = 0; t < 4; ++t) acc[mt][t] = (v8f){0, 0, 0, 0, 0, 0, 0, 0};

  // pipeline stage 0
  v16bf a0 = load_a_frag(xb, D_, m0, 0, lane);
  v16bf a1 = load_a_frag(xb, D_, m0 + 16, 0, lane);
  v16bf bv[4];
#pragma unroll
  for (int t = 0; t < 4; ++t) bv[t] = load_b_frag(wb, D_, n0 + 16 * t, 0, lane);

#pragma unroll 2
  for (int k0 = 32; k0 < D_; k0 += 32) {
    // issue next-stage loads first so they overlap the WMMAs below
    v16bf na0 = load_a_frag(xb, D_, m0, k0, lane);
    v16bf na1 = load_a_frag(xb, D_, m0 + 16, k0, lane);
    v16bf nb[4];
#pragma unroll
    for (int t = 0; t < 4; ++t)
      nb[t] = load_b_frag(wb, D_, n0 + 16 * t, k0, lane);
#pragma unroll
    for (int t = 0; t < 4; ++t) {
      acc[0][t] = wmma_bf16(a0, bv[t], acc[0][t]);
      acc[1][t] = wmma_bf16(a1, bv[t], acc[1][t]);
    }
    a0 = na0;
    a1 = na1;
#pragma unroll
    for (int t = 0; t < 4; ++t) bv[t] = nb[t];
  }
#pragma unroll
  for (int t = 0; t < 4; ++t) {
    acc[0][t] = wmma_bf16(a0, bv[t], acc[0][t]);
    acc[1][t] = wmma_bf16(a1, bv[t], acc[1][t]);
  }

  const int mlane = lane & 15, hg = lane >> 4;
#pragma unroll
  for (int mt = 0; mt < 2; ++mt) {
#pragma unroll
    for (int t = 0; t < 4; ++t) {
      int col = n0 + 16 * t + mlane;     // [0, 2304)
      int which = col / D_;              // 0=q 1=k 2=v
      int within = col - which * D_;
      int h = within >> 6, d = within & 63;
      float bia = bias[col];
#pragma unroll
      for (int r = 0; r < 8; ++r) {
        int row = m0 + mt * 16 + r + 8 * hg;  // [0, 8192)
        int b = row >> 10, s = row & 1023;
        bf16_t bvv = f2bf(acc[mt][t][r] + bia);
        size_t bh = (size_t)b * H_ + h;
        if (which == 0)
          Q[(bh * S_ + s) * HD_ + d] = bvv;
        else if (which == 1)
          Kc[(bh * S_ + s) * HD_ + d] = bvv;
        else
          Vt[(bh * HD_ + d) * S_ + s] = bvv;
      }
    }
  }
}

// ---------------------------------------------------------------------------
// Flash attention (unscaled QK^T per reference). 1 WG = (b, h, 64-query blk),
// 4 waves x 16 queries. Key chunks of 64. P staged through per-wave LDS
// buffer to convert C-layout -> A-fragment layout (in-order LDS, no barrier).
// Next chunk's K rows / Vt segments prefetched while softmax+PV runs.
__global__ __launch_bounds__(128) void flash_attn(
    const bf16_t* __restrict__ Q, const bf16_t* __restrict__ Kc,
    const bf16_t* __restrict__ Vt, bf16_t* __restrict__ O) {
  __shared__ bf16_t Pbuf[4 * 16 * 64];
  const int lane = threadIdx.x & 31;
  const int wave = threadIdx.x >> 5;
  const int h = blockIdx.y, b = blockIdx.z;
  const int q0 = blockIdx.x * 64 + wave * 16;
  const size_t bh = (size_t)b * H_ + h;
  const bf16_t* Qp = Q + bh * (S_ * HD_);
  const bf16_t* Kp = Kc + bh * (S_ * HD_);
  const bf16_t* Vp = Vt + bh * (HD_ * S_);
  bf16_t* Pw = Pbuf + wave * (16 * 64);

  const int mlane = lane & 15, g = lane >> 4;

  // Q fragments held in registers for the whole pass (hd=64 -> 2 A-frags)
  v16bf qa0 = load_a_frag(Qp, HD_, q0, 0, lane);
  v16bf qa1 = load_a_frag(Qp, HD_, q0, 32, lane);

  v8f o[4];
#pragma unroll
  for (int t = 0; t < 4; ++t) o[t] = (v8f){0, 0, 0, 0, 0, 0, 0, 0};
  float mrow[8], lrow[8];
#pragma unroll
  for (int r = 0; r < 8; ++r) {
    mrow[r] = -__builtin_inff();
    lrow[r] = 0.0f;
  }

  for (int kb = 0; kb < S_; kb += 64) {
    // ---- scores: 16 queries x 64 keys (4 n-tiles, K-dim 64 = 2 WMMAs each)
    v8f s[4];
#pragma unroll
    for (int t = 0; t < 4; ++t) {
      v16bf bk0 = load_b_frag(Kp, HD_, kb + 16 * t, 0, lane);
      v16bf bk1 = load_b_frag(Kp, HD_, kb + 16 * t, 32, lane);
      v8f z = (v8f){0, 0, 0, 0, 0, 0, 0, 0};
      z = wmma_bf16(qa0, bk0, z);
      z = wmma_bf16(qa1, bk1, z);
      s[t] = z;
    }

    // ---- prefetch next chunk (K rows are 128B each; Vt segs 128B each):
    // lanes cover 64 rows with 2 prefetches apiece -> global_prefetch_b8
    int kn = kb + 64;
    if (kn < S_) {
      __builtin_prefetch(Kp + (size_t)(kn + lane) * HD_, 0, 1);
      __builtin_prefetch(Kp + (size_t)(kn + 32 + lane) * HD_, 0, 1);
      __builtin_prefetch(Vp + (size_t)(lane & 31) * S_ + kn, 0, 1);
      __builtin_prefetch(Vp + (size_t)((lane & 31) + 32) * S_ + kn, 0, 1);
    }

    // ---- online softmax: row = r + 8g across the 16 lanes of a half-wave
    float mnew[8], scale[8];
#pragma unroll
    for (int r = 0; r < 8; ++r) {
      float mx = fmaxf(fmaxf(s[0][r], s[1][r]), fmaxf(s[2][r], s[3][r]));
#pragma unroll
      for (int off = 1; off < 16; off <<= 1)
        mx = fmaxf(mx, __shfl_xor(mx, off, 32));
      mnew[r] = fmaxf(mrow[r], mx);
      scale[r] = __expf(mrow[r] - mnew[r]);
      mrow[r] = mnew[r];
    }
    // ---- probabilities -> bf16 into LDS (row-major 16x64 per wave)
#pragma unroll
    for (int t = 0; t < 4; ++t) {
#pragma unroll
      for (int r = 0; r < 8; ++r) {
        float p = __expf(s[t][r] - mnew[r]);
        s[t][r] = p;
        Pw[(r + 8 * g) * 64 + 16 * t + mlane] = f2bf(p);
      }
    }
#pragma unroll
    for (int r = 0; r < 8; ++r) {
      float rs = (s[0][r] + s[1][r]) + (s[2][r] + s[3][r]);
#pragma unroll
      for (int off = 1; off < 16; off <<= 1) rs += __shfl_xor(rs, off, 32);
      lrow[r] = lrow[r] * scale[r] + rs;
    }
#pragma unroll
    for (int t = 0; t < 4; ++t)
#pragma unroll
      for (int r = 0; r < 8; ++r) o[t][r] *= scale[r];

    // ---- O += P @ V : P A-frags from LDS, V B-frags contiguous from Vt
#pragma unroll
    for (int kk = 0; kk < 2; ++kk) {
      v16bf pa = load_a_frag(Pw, 64, 0, 32 * kk, lane);
#pragma unroll
      for (int t = 0; t < 4; ++t) {
        v16bf bvv = load_b_frag(Vp, S_, 16 * t, kb + 32 * kk, lane);
        o[t] = wmma_bf16(pa, bvv, o[t]);
      }
    }
  }

  // ---- finalize and write attn output as bf16 [B, S, H*hd]
#pragma unroll
  for (int t = 0; t < 4; ++t) {
#pragma unroll
    for (int r = 0; r < 8; ++r) {
      int row = q0 + r + 8 * g;
      int col = h * HD_ + 16 * t + mlane;
      O[((size_t)b * S_ + row) * D_ + col] = f2bf(o[t][r] / lrow[r]);
    }
  }
}

// ---------------------------------------------------------------------------
// out = attn @ w_proj^T + b_proj, fp32 output. Grid (768/128=6, 8192/128=64).
// Same software-pipelined k-loop as qkv_gemm (unroll x2, no rotation moves).
__global__ __launch_bounds__(256) void proj_gemm(
    const bf16_t* __restrict__ ab,   // [M_, D_]
    const bf16_t* __restrict__ wb,   // [D_, D_]
    const float* __restrict__ bias,  // [D_]
    float* __restrict__ out) {       // [M_, D_]
  const int lane = threadIdx.x & 31;
  const int wave = threadIdx.x >> 5;
  const int m0 = blockIdx.y * 128 + (wave & 3) * 32;
  const int n0 = blockIdx.x * 128 + (wave >> 2) * 64;

  v8f acc[2][4];
#pragma unroll
  for (int mt = 0; mt < 2; ++mt)
#pragma unroll
    for (int t = 0; t < 4; ++t) acc[mt][t] = (v8f){0, 0, 0, 0, 0, 0, 0, 0};

  v16bf a0 = load_a_frag(ab, D_, m0, 0, lane);
  v16bf a1 = load_a_frag(ab, D_, m0 + 16, 0, lane);
  v16bf bv[4];
#pragma unroll
  for (int t = 0; t < 4; ++t) bv[t] = load_b_frag(wb, D_, n0 + 16 * t, 0, lane);

#pragma unroll 2
  for (int k0 = 32; k0 < D_; k0 += 32) {
    v16bf na0 = load_a_frag(ab, D_, m0, k0, lane);
    v16bf na1 = load_a_frag(ab, D_, m0 + 16, k0, lane);
    v16bf nb[4];
#pragma unroll
    for (int t = 0; t < 4; ++t)
      nb[t] = load_b_frag(wb, D_, n0 + 16 * t, k0, lane);
#pragma unroll
    for (int t = 0; t < 4; ++t) {
      acc[0][t] = wmma_bf16(a0, bv[t], acc[0][t]);
      acc[1][t] = wmma_bf16(a1, bv[t], acc[1][t]);
    }
    a0 = na0;
    a1 = na1;
#pragma unroll
    for (int t = 0; t < 4; ++t) bv[t] = nb[t];
  }
#pragma unroll
  for (int t = 0; t < 4; ++t) {
    acc[0][t] = wmma_bf16(a0, bv[t], acc[0][t]);
    acc[1][t] = wmma_bf16(a1, bv[t], acc[1][t]);
  }

  const int mlane = lane & 15, hg = lane >> 4;
#pragma unroll
  for (int mt = 0; mt < 2; ++mt) {
#pragma unroll
    for (int t = 0; t < 4; ++t) {
      int col = n0 + 16 * t + mlane;
      float bia = bias[col];
#pragma unroll
      for (int r = 0; r < 8; ++r) {
        int row = m0 + mt * 16 + r + 8 * hg;
        out[(size_t)row * D_ + col] = acc[mt][t][r] + bia;
      }
    }
  }
}

// ---------------------------------------------------------------------------
extern "C" void kernel_launch(void* const* d_in, const int* in_sizes, int n_in,
                              void* d_out, int out_size, void* d_ws,
                              size_t ws_size, hipStream_t stream) {
  const float* x = (const float*)d_in[0];       // [8,1024,768]
  const float* w_qkv = (const float*)d_in[1];   // [2304,768]
  const float* b_qkv = (const float*)d_in[2];   // [2304]
  const float* w_proj = (const float*)d_in[3];  // [768,768]
  const float* b_proj = (const float*)d_in[4];  // [768]
  float* out = (float*)d_out;                   // [8,1024,768]

  char* ws = (char*)d_ws;
  size_t off = 0;
  auto alloc = [&](size_t bytes) -> char* {
    char* p = ws + off;
    off += (bytes + 255) & ~(size_t)255;
    return p;
  };
  bf16_t* xb = (bf16_t*)alloc((size_t)M_ * D_ * 2);      // 12.6 MB
  bf16_t* wqkvb = (bf16_t*)alloc((size_t)N3D_ * D_ * 2); // 3.5 MB
  bf16_t* wprojb = (bf16_t*)alloc((size_t)D_ * D_ * 2);  // 1.2 MB
  bf16_t* Qb = (bf16_t*)alloc((size_t)M_ * D_ * 2);      // 12.6 MB
  bf16_t* Kb = (bf16_t*)alloc((size_t)M_ * D_ * 2);      // 12.6 MB
  bf16_t* Vtb = (bf16_t*)alloc((size_t)M_ * D_ * 2);     // 12.6 MB
  bf16_t* attnb = (bf16_t*)alloc((size_t)M_ * D_ * 2);   // 12.6 MB  (~68 MB total)

  cvt_f32_bf16<<<1024, 256, 0, stream>>>(x, xb, M_ * D_);
  cvt_f32_bf16<<<512, 256, 0, stream>>>(w_qkv, wqkvb, N3D_ * D_);
  cvt_f32_bf16<<<256, 256, 0, stream>>>(w_proj, wprojb, D_ * D_);

  qkv_gemm<<<dim3(N3D_ / 128, M_ / 128), 256, 0, stream>>>(
      xb, wqkvb, b_qkv, Qb, Kb, Vtb);

  flash_attn<<<dim3(S_ / 64, H_, B_), 128, 0, stream>>>(Qb, Kb, Vtb, attnb);

  proj_gemm<<<dim3(D_ / 128, M_ / 128), 256, 0, stream>>>(
      attnb, wprojb, b_proj, out);
}
